// CRF_18571438588060
// MI455X (gfx1250) — matrix operand order repaired and checked by reading, not compiled, not run on hardware
//
#include <hip/hip_runtime.h>

typedef __attribute__((ext_vector_type(16))) _Float16 v16h;
typedef __attribute__((ext_vector_type(8)))  float    v8f;

#define CRF_NT     32
#define CRF_L      16384
#define CRF_P      256                 // stage-1 parallel chunks
#define CRF_S      (CRF_L / CRF_P)     // 64 timesteps per chunk
#define CRF_G2     8                   // stage-2 fold factor
#define CRF_P2     (CRF_P / CRF_G2)    // 32 stage-2 outputs
#define CRF_START  0
#define CRF_END    1
#define CRF_FLOOR  (-1.0e30f)          // finite "log zero"

// ---------------------------------------------------------------------------
// WMMA helper: D = A(16x32 f16) * B(32x16 f16) + C(16x16 f32), wave32
// ---------------------------------------------------------------------------
static __device__ __forceinline__ v8f crf_wmma(v16h a, v16h b, v8f c) {
  return __builtin_amdgcn_wmma_f32_16x16x32_f16(
      /*neg_a=*/false, a, /*neg_b=*/false, b,
      /*c_mod=*/(short)0, c, /*reuse_a=*/false, /*reuse_b=*/false);
}

// A fragment: 16x32 f16 tile, rows [16*mt, 16*mt+16).
// K order used here (contiguous, same as B): lane l holds row M=l%16; VGPR v, half h:
//   K = v*2 + h + (l/16)*16
static __device__ __forceinline__ v16h
crf_load_a_frag(const _Float16 (*Pm)[CRF_NT], int mt, int lane) {
  v16h a = {};
  const int M  = (mt << 4) + (lane & 15);
  const int hk = ((lane >> 4) & 1) << 4;
#pragma unroll
  for (int v = 0; v < 8; ++v) {
    const int k = (v << 1) + hk;
    a[2 * v]     = Pm[M][k];
    a[2 * v + 1] = Pm[M][k + 1];
  }
  return a;
}

// B fragment: 32x16 f16 tile, cols [16*nt, 16*nt+16).
// Contiguous k order used here, same as A: lanes 0-15 hold K=0..15
// (VGPR v half h -> K=2v+h), lanes 16-31 hold K=16..31.
static __device__ __forceinline__ v16h
crf_load_b_frag(const _Float16 (*Bm)[CRF_NT], int nt, int lane) {
  v16h b = {};
  const int N  = (nt << 4) + (lane & 15);
  const int kb = ((lane >> 4) & 1) << 4;
#pragma unroll
  for (int v = 0; v < 8; ++v) {
    const int k = kb + (v << 1);
    b[2 * v]     = Bm[k][N];
    b[2 * v + 1] = Bm[k + 1][N];
  }
  return b;
}

// ---------------------------------------------------------------------------
// In-place stabilized log-semiring combine:  R = R (x) B
//   R[i][j] <- mA_i + nB_j + log( sum_k exp(R[i][k]-mA_i) * exp(B[k][j]-nB_j) )
// Called by ALL threads of the block (barriers are uniform); memory work and
// WMMA are guarded by `active`, which is uniform per wave (EXEC all-ones).
// R must be LDS-backed; B may be global or LDS (generic pointer).
// ---------------------------------------------------------------------------
static __device__ __forceinline__ void crf_combine_inplace(
    bool active, int lane, float* __restrict__ R, const float* __restrict__ B,
    _Float16 (*sEA)[CRF_NT], _Float16 (*sEB)[CRF_NT],
    float* sMA, float* sNB) {
  if (active) {
    // Lane i: row i of R, row-max shift.
    float m = R[lane * CRF_NT];
#pragma unroll
    for (int k = 1; k < CRF_NT; ++k) m = fmaxf(m, R[lane * CRF_NT + k]);
    sMA[lane] = m;
#pragma unroll
    for (int k = 0; k < CRF_NT; ++k)
      sEA[lane][k] = (_Float16)__expf(R[lane * CRF_NT + k] - m);
    // Lane j: column j of B, col-max shift.
    float n = B[lane];
#pragma unroll
    for (int k = 1; k < CRF_NT; ++k) n = fmaxf(n, B[k * CRF_NT + lane]);
    sNB[lane] = n;
#pragma unroll
    for (int k = 0; k < CRF_NT; ++k)
      sEB[k][lane] = (_Float16)__expf(B[k * CRF_NT + lane] - n);
  }
  __syncthreads();
  if (active) {
    const v16h a0 = crf_load_a_frag(sEA, 0, lane);
    const v16h a1 = crf_load_a_frag(sEA, 1, lane);
    const v16h b0 = crf_load_b_frag(sEB, 0, lane);
    const v16h b1 = crf_load_b_frag(sEB, 1, lane);
    v8f c00 = {}, c01 = {}, c10 = {}, c11 = {};
    c00 = crf_wmma(a0, b0, c00);
    c01 = crf_wmma(a0, b1, c01);
    c10 = crf_wmma(a1, b0, c10);
    c11 = crf_wmma(a1, b1, c11);

    const int nn = lane & 15;
    const int hi = (lane >> 4) & 1;
#pragma unroll
    for (int r = 0; r < 8; ++r) {
      const int m0 = r + hi * 8;
      const int m1 = 16 + r + hi * 8;
      float q;
      q = c00[r];
      R[m0 * CRF_NT + nn]      = fmaxf(sMA[m0] + sNB[nn]      + (q > 0.f ? __logf(q) : CRF_FLOOR), CRF_FLOOR);
      q = c01[r];
      R[m0 * CRF_NT + nn + 16] = fmaxf(sMA[m0] + sNB[nn + 16] + (q > 0.f ? __logf(q) : CRF_FLOOR), CRF_FLOOR);
      q = c10[r];
      R[m1 * CRF_NT + nn]      = fmaxf(sMA[m1] + sNB[nn]      + (q > 0.f ? __logf(q) : CRF_FLOOR), CRF_FLOOR);
      q = c11[r];
      R[m1 * CRF_NT + nn + 16] = fmaxf(sMA[m1] + sNB[nn + 16] + (q > 0.f ? __logf(q) : CRF_FLOOR), CRF_FLOOR);
    }
  }
  __syncthreads();
}

// ---------------------------------------------------------------------------
// Kernel 1: each single-wave workgroup folds S=64 timesteps into one 32x32
// chunk matrix, working in the EXP domain with per-row running log-scales:
//   P = Q / rowmax(Q);  mlog += log(rowmax);  Q' = (P x expT) .* expE_t
// Only ~2 transcendentals per lane per step; sums ride the WMMA pipe.
// ---------------------------------------------------------------------------
__global__ __launch_bounds__(32) void crf_chunk_kernel(
    const int* __restrict__ x, const float* __restrict__ emit,
    const float* __restrict__ trans, float* __restrict__ out) {
  __shared__ float    sQ[CRF_NT][CRF_NT];   // exp-domain running matrix
  __shared__ _Float16 sP[CRF_NT][CRF_NT];   // rescaled rows (WMMA A operand)
  __shared__ float    sE[CRF_S][CRF_NT];    // exp(emissions), prefetched

  const int lane = threadIdx.x;             // 0..31
  const int c    = blockIdx.x;
  const int t0   = c * CRF_S;
  const int nn   = lane & 15;
  const int hi   = (lane >> 4) & 1;

  // Prefetch + pre-exponentiate this chunk's emissions (gather from 256MB table).
  for (int t = 0; t < CRF_S; ++t) {
    const long long row = (long long)x[t0 + t] * CRF_NT;
    sE[t][lane] = __expf(emit[row + lane]);
  }

  // Constant B fragments: exp(T). Masked entries (-10000) become exactly 0.
  v16h b0 = {}, b1 = {};
  {
    const int kb = hi << 4;
#pragma unroll
    for (int v = 0; v < 8; ++v) {
      const int k = kb + (v << 1);
      b0[2 * v]     = (_Float16)__expf(trans[k * CRF_NT + nn]);
      b0[2 * v + 1] = (_Float16)__expf(trans[(k + 1) * CRF_NT + nn]);
      b1[2 * v]     = (_Float16)__expf(trans[k * CRF_NT + nn + 16]);
      b1[2 * v + 1] = (_Float16)__expf(trans[(k + 1) * CRF_NT + nn + 16]);
    }
  }

  // Exp-domain identity.
#pragma unroll
  for (int j = 0; j < CRF_NT; ++j)
    sQ[lane][j] = (lane == j) ? 1.0f : 0.0f;
  __syncthreads();

  float mlog = 0.0f;  // lane l carries log-scale of row l
  for (int t = 0; t < CRF_S; ++t) {
    // Lane l owns row l: rescale by row max, accumulate log-scale.
    float q[CRF_NT];
#pragma unroll
    for (int j = 0; j < CRF_NT; ++j) q[j] = sQ[lane][j];
    float m = q[0];
#pragma unroll
    for (int j = 1; j < CRF_NT; ++j) m = fmaxf(m, q[j]);
    m = fmaxf(m, 1.0e-37f);                    // dead rows stay all-zero
    const float inv = __builtin_amdgcn_rcpf(m);
    mlog += __logf(m);
#pragma unroll
    for (int j = 0; j < CRF_NT; ++j)
      sP[lane][j] = (_Float16)(q[j] * inv);
    __syncthreads();

    const v16h a0 = crf_load_a_frag(sP, 0, lane);
    const v16h a1 = crf_load_a_frag(sP, 1, lane);
    v8f c00 = {}, c01 = {}, c10 = {}, c11 = {};
    c00 = crf_wmma(a0, b0, c00);
    c01 = crf_wmma(a0, b1, c01);
    c10 = crf_wmma(a1, b0, c10);
    c11 = crf_wmma(a1, b1, c11);

    // Scatter Q' = C .* expE_t (emission folded in as a column multiply).
    // C layout: VGPR r, lane l -> M = r + (l/16)*8 (+16 per row-tile), N = l%16.
    const float e0 = sE[t][nn];
    const float e1 = sE[t][nn + 16];
#pragma unroll
    for (int r = 0; r < 8; ++r) {
      const int m0 = r + hi * 8;
      const int m1 = 16 + r + hi * 8;
      sQ[m0][nn]      = c00[r] * e0;
      sQ[m0][nn + 16] = c01[r] * e1;
      sQ[m1][nn]      = c10[r] * e0;
      sQ[m1][nn + 16] = c11[r] * e1;
    }
    __syncthreads();
  }

  // Back to log domain: out[l][j] = mlog_l + log(Q[l][j]).
#pragma unroll
  for (int j = 0; j < CRF_NT; ++j) {
    const float qv = sQ[lane][j];
    out[(size_t)c * (CRF_NT * CRF_NT) + lane * CRF_NT + j] =
        fmaxf(mlog + (qv > 0.f ? __logf(qv) : CRF_FLOOR), CRF_FLOOR);
  }
}

// ---------------------------------------------------------------------------
// Kernel 2: 32 single-wave blocks; block g left-folds chunk matrices
// [8g, 8g+8) into one matrix (order-preserving).
// ---------------------------------------------------------------------------
__global__ __launch_bounds__(32) void crf_fold8_kernel(
    const float* __restrict__ in, float* __restrict__ out) {
  __shared__ float    sR[CRF_NT][CRF_NT];
  __shared__ _Float16 sEA[CRF_NT][CRF_NT];
  __shared__ _Float16 sEB[CRF_NT][CRF_NT];
  __shared__ float    sMA[CRF_NT];
  __shared__ float    sNB[CRF_NT];

  const int    lane = threadIdx.x;
  const size_t g    = blockIdx.x;
  const float* base = in + g * (size_t)CRF_G2 * (CRF_NT * CRF_NT);

#pragma unroll
  for (int j = 0; j < CRF_NT; ++j) sR[lane][j] = base[lane * CRF_NT + j];
  __syncthreads();

  for (int r = 1; r < CRF_G2; ++r)
    crf_combine_inplace(true, lane, &sR[0][0],
                        base + (size_t)r * (CRF_NT * CRF_NT),
                        sEA, sEB, sMA, sNB);

#pragma unroll
  for (int j = 0; j < CRF_NT; ++j)
    out[g * (size_t)(CRF_NT * CRF_NT) + lane * CRF_NT + j] = sR[lane][j];
}

// ---------------------------------------------------------------------------
// Kernel 3: one block, 4 waves. Phase 1: wave w folds its 8 matrices into
// LDS slot R_w. Phase 2: 2-round in-LDS tree (R0=R0xR1, R2=R2xR3; R0=R0xR2).
// Finalize: z = lse_j( lse_i(init_i + Mtot[i][j]) + T[j][END] ) on wave 0.
// ---------------------------------------------------------------------------
__global__ __launch_bounds__(128) void crf_root_kernel(
    const float* __restrict__ in, const float* __restrict__ trans,
    float* __restrict__ out) {
  __shared__ float    sR[4][CRF_NT][CRF_NT];
  __shared__ _Float16 sEA[4][CRF_NT][CRF_NT];
  __shared__ _Float16 sEB[4][CRF_NT][CRF_NT];
  __shared__ float    sMA[4][CRF_NT];
  __shared__ float    sNB[4][CRF_NT];

  const int lane = threadIdx.x & 31;
  const int wv   = threadIdx.x >> 5;  // 0..3
  const float* base = in + (size_t)wv * CRF_G2 * (CRF_NT * CRF_NT);

#pragma unroll
  for (int j = 0; j < CRF_NT; ++j) sR[wv][lane][j] = base[lane * CRF_NT + j];
  __syncthreads();

  // Phase 1: each wave folds 8 global matrices into its LDS slot.
  for (int r = 1; r < CRF_G2; ++r)
    crf_combine_inplace(true, lane, &sR[wv][0][0],
                        base + (size_t)r * (CRF_NT * CRF_NT),
                        sEA[wv], sEB[wv], sMA[wv], sNB[wv]);

  // Phase 2 round 1: wave0: R0 <- R0 x R1 ; wave1: R2 <- R2 x R3.
  {
    const bool act = (wv < 2);
    const int  ri  = act ? (wv * 2) : 0;
    const int  bi  = act ? (wv * 2 + 1) : 0;
    crf_combine_inplace(act, lane, &sR[ri][0][0], &sR[bi][0][0],
                        sEA[wv], sEB[wv], sMA[wv], sNB[wv]);
  }
  // Phase 2 round 2: wave0: R0 <- R0 x R2.
  crf_combine_inplace(wv == 0, lane, &sR[0][0][0], &sR[2][0][0],
                      sEA[wv], sEB[wv], sMA[wv], sNB[wv]);

  // Finalize on wave 0 (matches reference init vector exactly).
  if (wv == 0) {
    float mx = -3.0e38f;
#pragma unroll
    for (int i = 0; i < CRF_NT; ++i) {
      const float v = ((i == CRF_START) ? 0.0f : -10000.0f) + sR[0][i][lane];
      mx = fmaxf(mx, v);
    }
    float s = 0.0f;
#pragma unroll
    for (int i = 0; i < CRF_NT; ++i) {
      const float v = ((i == CRF_START) ? 0.0f : -10000.0f) + sR[0][i][lane];
      s += __expf(v - mx);
    }
    const float zj = mx + __logf(s) + trans[lane * CRF_NT + CRF_END];

    // Wave32 logsumexp reduction via shuffles.
    float mz = zj;
#pragma unroll
    for (int off = 16; off > 0; off >>= 1)
      mz = fmaxf(mz, __shfl_xor(mz, off, 32));
    float sz = __expf(zj - mz);
#pragma unroll
    for (int off = 16; off > 0; off >>= 1)
      sz += __shfl_xor(sz, off, 32);
    if (lane == 0) out[0] = mz + __logf(sz);
  }
}

// ---------------------------------------------------------------------------
extern "C" void kernel_launch(void* const* d_in, const int* in_sizes, int n_in,
                              void* d_out, int out_size, void* d_ws, size_t ws_size,
                              hipStream_t stream) {
  (void)in_sizes; (void)n_in; (void)out_size; (void)ws_size;
  const int*   x     = (const int*)d_in[0];
  const float* emit  = (const float*)d_in[1];
  const float* trans = (const float*)d_in[2];

  float* buf1 = (float*)d_ws;                                  // 256 * 4KB
  float* buf2 = buf1 + (size_t)CRF_P * CRF_NT * CRF_NT;        // 32 * 4KB

  crf_chunk_kernel<<<CRF_P, 32, 0, stream>>>(x, emit, trans, buf1);
  crf_fold8_kernel<<<CRF_P2, 32, 0, stream>>>(buf1, buf2);
  crf_root_kernel<<<1, 128, 0, stream>>>(buf2, trans, (float*)d_out);
}
